// PointNetSetAbstraction_17824114278740
// MI455X (gfx1250) — matrix operand order repaired and checked by reading, not compiled, not run on hardware
//
#include <hip/hip_runtime.h>
#include <cstdint>

typedef __attribute__((ext_vector_type(16))) _Float16 v16h;
typedef __attribute__((ext_vector_type(8)))  _Float16 v8h;
typedef __attribute__((ext_vector_type(8)))  float    v8f;

constexpr int NB   = 16;    // batch
constexpr int NPTS = 8192;  // N
constexpr int CIN  = 64;
constexpr int NM   = 1024;  // npoint
constexpr int KNN  = 32;
constexpr int HID  = 64;
constexpr int OUTC = 128;
constexpr float EPSV = 1e-5f;

constexpr int W1PAD = 64 * 96;    // padded f16 image of w1 (K padded 67->96)
constexpr int W2PAD = 128 * 64;   // f16 image of w2

// ---------------------------------------------------------------------------
// CDNA5 async global->LDS copy (ASYNCcnt-tracked), GV addressing form.
// Generic pointers to LDS truncate to the wave-relative LDS offset (low 32
// bits of the address), which is what VDST expects.
// ---------------------------------------------------------------------------
__device__ __forceinline__ void async_copy_b128(void* lds_dst, const void* gsrc) {
    uint32_t l = (uint32_t)(uintptr_t)lds_dst;
    uint64_t g = (uint64_t)(uintptr_t)gsrc;
    asm volatile("global_load_async_to_lds_b128 %0, %1, off"
                 :: "v"(l), "v"(g) : "memory");
}
__device__ __forceinline__ void wait_async0() {
    asm volatile("s_wait_asynccnt 0" ::: "memory");
}

// ---------------------------------------------------------------------------
// one-time: convert weights to padded f16 images for raw async staging
// ---------------------------------------------------------------------------
__global__ __launch_bounds__(256) void prep_w_kernel(const float* __restrict__ w1,
                                                     const float* __restrict__ w2,
                                                     _Float16* __restrict__ w1h,
                                                     _Float16* __restrict__ w2h)
{
    int i = blockIdx.x * 256 + threadIdx.x;
    if (i < W1PAD) {
        int oc = i / 96, c = i % 96;
        w1h[i] = (_Float16)((c < 67) ? w1[oc * 67 + c] : 0.0f);
    } else if (i < W1PAD + W2PAD) {
        int k = i - W1PAD;
        w2h[k] = (_Float16)w2[k];
    }
}

// ---------------------------------------------------------------------------
// FPS: one block per batch. 1024 threads, 8 points/thread held in VGPRs.
// ---------------------------------------------------------------------------
__global__ __launch_bounds__(1024) void fps_kernel(const float* __restrict__ xyz,
                                                   int* __restrict__ fps_idx,
                                                   float* __restrict__ cent,
                                                   float* __restrict__ out_cent)
{
    const int b = blockIdx.x;
    const int t = threadIdx.x;
    const float* xb = xyz + (size_t)b * NPTS * 3;

    float px[8], py[8], pz[8], dist[8];
#pragma unroll
    for (int i = 0; i < 8; ++i) {
        int n = t + i * 1024;
        px[i] = xb[n * 3 + 0];
        py[i] = xb[n * 3 + 1];
        pz[i] = xb[n * 3 + 2];
        dist[i] = 1e10f;
    }

    __shared__ float scx, scy, scz;
    __shared__ float rd[32];
    __shared__ int   ri[32];
    __shared__ int   sfar;

    if (t == 0) {
        scx = px[0]; scy = py[0]; scz = pz[0];
        fps_idx[b * NM] = 0;
        cent[(size_t)(b * NM) * 3 + 0] = px[0];
        cent[(size_t)(b * NM) * 3 + 1] = py[0];
        cent[(size_t)(b * NM) * 3 + 2] = pz[0];
        out_cent[(size_t)(b * NM) * 3 + 0] = px[0];
        out_cent[(size_t)(b * NM) * 3 + 1] = py[0];
        out_cent[(size_t)(b * NM) * 3 + 2] = pz[0];
    }
    __syncthreads();

    for (int it = 0; it < NM - 1; ++it) {
        float cx = scx, cy = scy, cz = scz;
        float bd = -1.0f; int bi = 0x7fffffff;
#pragma unroll
        for (int i = 0; i < 8; ++i) {
            float dx = px[i] - cx, dy = py[i] - cy, dz = pz[i] - cz;
            float d  = dx * dx + dy * dy + dz * dz;
            float nd = fminf(dist[i], d);
            dist[i]  = nd;
            int n = t + i * 1024;
            if (nd > bd || (nd == bd && n < bi)) { bd = nd; bi = n; }
        }
#pragma unroll
        for (int off = 16; off > 0; off >>= 1) {
            float od = __shfl_xor(bd, off, 32);
            int   oi = __shfl_xor(bi, off, 32);
            if (od > bd || (od == bd && oi < bi)) { bd = od; bi = oi; }
        }
        if ((t & 31) == 0) { rd[t >> 5] = bd; ri[t >> 5] = bi; }
        __syncthreads();
        if (t < 32) {
            bd = rd[t]; bi = ri[t];
#pragma unroll
            for (int off = 16; off > 0; off >>= 1) {
                float od = __shfl_xor(bd, off, 32);
                int   oi = __shfl_xor(bi, off, 32);
                if (od > bd || (od == bd && oi < bi)) { bd = od; bi = oi; }
            }
            if (t == 0) sfar = bi;
        }
        __syncthreads();
        int far = sfar;
        if ((far & 1023) == t) {
            int slot = far >> 10;
            scx = px[slot]; scy = py[slot]; scz = pz[slot];
            size_t o = (size_t)(b * NM + it + 1);
            fps_idx[o] = far;
            cent[o * 3 + 0] = px[slot];
            cent[o * 3 + 1] = py[slot];
            cent[o * 3 + 2] = pz[slot];
            out_cent[o * 3 + 0] = px[slot];
            out_cent[o * 3 + 1] = py[slot];
            out_cent[o * 3 + 2] = pz[slot];
        }
        __syncthreads();
    }
}

// ---------------------------------------------------------------------------
// kNN: one block per (b,m), register-resident d2, 32 extraction rounds.
// ---------------------------------------------------------------------------
__global__ __launch_bounds__(256) void knn_kernel(const float* __restrict__ xyz,
                                                  const float* __restrict__ cent,
                                                  int* __restrict__ knn_idx)
{
    const int bm = blockIdx.x;
    const int b = bm >> 10;
    const int t = threadIdx.x;
    const float* xb = xyz + (size_t)b * NPTS * 3;
    const float cx = cent[(size_t)bm * 3 + 0];
    const float cy = cent[(size_t)bm * 3 + 1];
    const float cz = cent[(size_t)bm * 3 + 2];

    float d2[32];
#pragma unroll
    for (int i = 0; i < 32; ++i) {
        int n = t + i * 256;
        float dx = xb[n * 3 + 0] - cx;
        float dy = xb[n * 3 + 1] - cy;
        float dz = xb[n * 3 + 2] - cz;
        d2[i] = dx * dx + dy * dy + dz * dz;
    }

    __shared__ float rd[8];
    __shared__ int   ri[8];
    __shared__ int   sbi;

    for (int r = 0; r < KNN; ++r) {
        float bd = 3.4e38f; int bi = 0x7fffffff;
#pragma unroll
        for (int i = 0; i < 32; ++i) {
            int n = t + i * 256;
            if (d2[i] < bd || (d2[i] == bd && n < bi)) { bd = d2[i]; bi = n; }
        }
#pragma unroll
        for (int off = 16; off > 0; off >>= 1) {
            float od = __shfl_xor(bd, off, 32);
            int   oi = __shfl_xor(bi, off, 32);
            if (od < bd || (od == bd && oi < bi)) { bd = od; bi = oi; }
        }
        if ((t & 31) == 0) { rd[t >> 5] = bd; ri[t >> 5] = bi; }
        __syncthreads();
        if (t == 0) {
            float xd = rd[0]; int xi = ri[0];
            for (int wv = 1; wv < 8; ++wv)
                if (rd[wv] < xd || (rd[wv] == xd && ri[wv] < xi)) { xd = rd[wv]; xi = ri[wv]; }
            sbi = xi;
            knn_idx[(size_t)bm * KNN + r] = xi;
        }
        __syncthreads();
        int win = sbi;
        if ((win & 255) == t) d2[win >> 8] = 3.4e38f;
        __syncthreads();
    }
}

// ---------------------------------------------------------------------------
// conv1: async weight staging + gather + WMMA (64x67)x(67x32) per (b,m).
// ---------------------------------------------------------------------------
__global__ __launch_bounds__(256) void conv1_kernel(const float* __restrict__ xyz,
                                                    const float* __restrict__ feat,
                                                    const float* __restrict__ cent,
                                                    const int* __restrict__ knn_idx,
                                                    const _Float16* __restrict__ w1h,
                                                    const float* __restrict__ b1,
                                                    _Float16* __restrict__ x1h,
                                                    float* __restrict__ ps1)
{
    __shared__ __align__(16) _Float16 Xs[32 * 96];   // [pixel][channel], padded
    __shared__ __align__(16) _Float16 Ws[W1PAD];     // [outch][channel]
    __shared__ float Os[64 * 32];
    __shared__ float gsum[64], gsq[64];

    const int bm = blockIdx.x;
    const int b = bm >> 10;
    const int m = bm & 1023;
    const int t = threadIdx.x;

    // async global->LDS weight staging (3 x b128 per thread, ASYNCcnt)
#pragma unroll
    for (int k = 0; k < 3; ++k) {
        int ch = t + k * 256;                 // 768 chunks of 8 f16
        async_copy_b128(&Ws[ch * 8], &w1h[ch * 8]);
    }

    // overlapped: gather the X tile (f32 -> f16, with delta channels)
    const float cx = cent[(size_t)bm * 3 + 0];
    const float cy = cent[(size_t)bm * 3 + 1];
    const float cz = cent[(size_t)bm * 3 + 2];
    {
        int jp = t >> 3;          // pixel 0..31
        int l8 = t & 7;           // 8 threads per pixel, 12 channels each
        int n  = knn_idx[(size_t)bm * KNN + jp];
        const float* xr = xyz  + ((size_t)b * NPTS + n) * 3;
        const float* fr = feat + ((size_t)b * NPTS + n) * CIN;
        for (int c = l8 * 12; c < l8 * 12 + 12; ++c) {
            float v;
            if (c < 3)       v = xr[c] - (c == 0 ? cx : (c == 1 ? cy : cz));
            else if (c < 67) v = fr[c - 3];
            else             v = 0.0f;
            Xs[jp * 96 + c] = (_Float16)v;
        }
    }
    wait_async0();
    __syncthreads();

    const int w = t >> 5, lane = t & 31;
    const int ocb = (w >> 1) * 16;
    const int pxb = (w & 1) * 16;
    const int Mrow = ocb + (lane & 15);
    const int col  = pxb + (lane & 15);
    const int khalf = (lane < 16) ? 0 : 8;
    const int kb    = (lane < 16) ? 0 : 16;

    v8f acc = {};
#pragma unroll
    for (int kc = 0; kc < 96; kc += 32) {
        v8h a0  = *(const v8h*)&Ws[Mrow * 96 + kc + khalf];
        v8h a1  = *(const v8h*)&Ws[Mrow * 96 + kc + 16 + khalf];
        v8h b0  = *(const v8h*)&Xs[col * 96 + kc + khalf];
        v8h b1v = *(const v8h*)&Xs[col * 96 + kc + 16 + khalf];
        v16h A, Bf;
#pragma unroll
        for (int e = 0; e < 8; ++e) { A[e] = a0[e]; A[e + 8] = a1[e]; Bf[e] = b0[e]; Bf[e + 8] = b1v[e]; }
        acc = __builtin_amdgcn_wmma_f32_16x16x32_f16(false, A, false, Bf, (short)0, acc, false, false);
    }
#pragma unroll
    for (int r = 0; r < 8; ++r) {
        int oc = ocb + r + ((lane < 16) ? 0 : 8);
        int j  = pxb + (lane & 15);
        Os[oc * 32 + j] = acc[r] + b1[oc];
    }
    __syncthreads();

    {   // vectorized f16 spill: one b128 store per thread
        int oc = t >> 2;
        int j0 = (t & 3) * 8;
        v8h pk;
#pragma unroll
        for (int e = 0; e < 8; ++e) pk[e] = (_Float16)Os[oc * 32 + j0 + e];
        *(v8h*)&x1h[((size_t)(b * HID + oc) * NM + m) * KNN + j0] = pk;
    }
    if (t < 64) {
        float S = 0.0f, Q = 0.0f;
        for (int j = 0; j < 32; ++j) { float v = Os[t * 32 + j]; S += v; Q += v * v; }
        gsum[t] = S; gsq[t] = Q;
    }
    __syncthreads();
    if (t < 32) {
        float S = gsum[2 * t] + gsum[2 * t + 1];
        float Q = gsq[2 * t]  + gsq[2 * t + 1];
        size_t o = (((size_t)b * 32 + t) * NM + m) * 2;
        ps1[o + 0] = S; ps1[o + 1] = Q;
    }
}

// ---------------------------------------------------------------------------
// conv2: async weight staging, GN1+ReLU fused into vectorized X load, WMMA.
// ---------------------------------------------------------------------------
__global__ __launch_bounds__(256) void conv2_kernel(const _Float16* __restrict__ x1h,
                                                    const float* __restrict__ mr1,
                                                    const float* __restrict__ g1w,
                                                    const float* __restrict__ g1b,
                                                    const _Float16* __restrict__ w2h,
                                                    const float* __restrict__ b2,
                                                    _Float16* __restrict__ x2h,
                                                    float* __restrict__ ps2)
{
    __shared__ __align__(16) _Float16 Xs[32 * 64];    // [pixel][channel]
    __shared__ __align__(16) _Float16 Ws[W2PAD];      // [outch][channel]
    __shared__ float Os[128 * 32];
    __shared__ float gsum[128], gsq[128];

    const int bm = blockIdx.x;
    const int b = bm >> 10;
    const int m = bm & 1023;
    const int t = threadIdx.x;

#pragma unroll
    for (int k = 0; k < 4; ++k) {
        int ch = t + k * 256;                 // 1024 chunks of 8 f16
        async_copy_b128(&Ws[ch * 8], &w2h[ch * 8]);
    }

    {   // coalesced v8h load of x1 (contiguous j-run), GN1+ReLU, scatter to LDS
        int c  = t >> 2;              // channel 0..63
        int j0 = (t & 3) * 8;         // j octet
        int g  = c >> 1;
        float mean = mr1[(b * 32 + g) * 2 + 0];
        float rstd = mr1[(b * 32 + g) * 2 + 1];
        float gw = g1w[c], gb = g1b[c];
        v8h raw = *(const v8h*)&x1h[((size_t)(b * HID + c) * NM + m) * KNN + j0];
#pragma unroll
        for (int e = 0; e < 8; ++e) {
            float v = ((float)raw[e] - mean) * rstd * gw + gb;
            Xs[(j0 + e) * 64 + c] = (_Float16)fmaxf(v, 0.0f);
        }
    }
    wait_async0();
    __syncthreads();

    const int w = t >> 5, lane = t & 31;
    const int khalf = (lane < 16) ? 0 : 8;
    const int kb    = (lane < 16) ? 0 : 16;

    for (int T = w; T < 16; T += 8) {
        int ocb = (T >> 1) * 16;
        int pxb = (T & 1) * 16;
        int Mrow = ocb + (lane & 15);
        int col  = pxb + (lane & 15);
        v8f acc = {};
#pragma unroll
        for (int kc = 0; kc < 64; kc += 32) {
            v8h a0  = *(const v8h*)&Ws[Mrow * 64 + kc + khalf];
            v8h a1  = *(const v8h*)&Ws[Mrow * 64 + kc + 16 + khalf];
            v8h b0  = *(const v8h*)&Xs[col * 64 + kc + khalf];
            v8h b1v = *(const v8h*)&Xs[col * 64 + kc + 16 + khalf];
            v16h A, Bf;
#pragma unroll
            for (int e = 0; e < 8; ++e) { A[e] = a0[e]; A[e + 8] = a1[e]; Bf[e] = b0[e]; Bf[e + 8] = b1v[e]; }
            acc = __builtin_amdgcn_wmma_f32_16x16x32_f16(false, A, false, Bf, (short)0, acc, false, false);
        }
#pragma unroll
        for (int r = 0; r < 8; ++r) {
            int oc = ocb + r + ((lane < 16) ? 0 : 8);
            int j  = pxb + (lane & 15);
            Os[oc * 32 + j] = acc[r] + b2[oc];
        }
    }
    __syncthreads();

#pragma unroll
    for (int k = 0; k < 2; ++k) {     // 512 b128 stores
        int ch = t + k * 256;
        int oc = ch >> 2;
        int j0 = (ch & 3) * 8;
        v8h pk;
#pragma unroll
        for (int e = 0; e < 8; ++e) pk[e] = (_Float16)Os[oc * 32 + j0 + e];
        *(v8h*)&x2h[((size_t)(b * OUTC + oc) * NM + m) * KNN + j0] = pk;
    }
    if (t < 128) {
        float S = 0.0f, Q = 0.0f;
        for (int j = 0; j < 32; ++j) { float v = Os[t * 32 + j]; S += v; Q += v * v; }
        gsum[t] = S; gsq[t] = Q;
    }
    __syncthreads();
    if (t < 32) {
        float S = gsum[4 * t] + gsum[4 * t + 1] + gsum[4 * t + 2] + gsum[4 * t + 3];
        float Q = gsq[4 * t]  + gsq[4 * t + 1]  + gsq[4 * t + 2]  + gsq[4 * t + 3];
        size_t o = (((size_t)b * 32 + t) * NM + m) * 2;
        ps2[o + 0] = S; ps2[o + 1] = Q;
    }
}

// ---------------------------------------------------------------------------
// finalize GN stats: deterministic tree reduction per (b,group).
// ---------------------------------------------------------------------------
__global__ __launch_bounds__(256) void finstats_kernel(const float* __restrict__ ps,
                                                       float* __restrict__ mr,
                                                       float inv_cnt)
{
    const int bg = blockIdx.x;
    const int t = threadIdx.x;
    __shared__ float sS[256], sQ[256];
    float S = 0.0f, Q = 0.0f;
    for (int m = t; m < NM; m += 256) {
        size_t o = ((size_t)bg * NM + m) * 2;
        S += ps[o + 0]; Q += ps[o + 1];
    }
    sS[t] = S; sQ[t] = Q;
    __syncthreads();
    for (int off = 128; off > 0; off >>= 1) {
        if (t < off) { sS[t] += sS[t + off]; sQ[t] += sQ[t + off]; }
        __syncthreads();
    }
    if (t == 0) {
        float mean = sS[0] * inv_cnt;
        float var  = sQ[0] * inv_cnt - mean * mean;
        mr[bg * 2 + 0] = mean;
        mr[bg * 2 + 1] = rsqrtf(var + EPSV);
    }
}

// ---------------------------------------------------------------------------
// final: GN2 + max + ReLU (monotone)  ->  new_feat (B,M,128)
// ---------------------------------------------------------------------------
__global__ __launch_bounds__(256) void final_kernel(const _Float16* __restrict__ x2h,
                                                    const float* __restrict__ mr2,
                                                    const float* __restrict__ g2w,
                                                    const float* __restrict__ g2b,
                                                    float* __restrict__ out_feat)
{
    size_t i = (size_t)blockIdx.x * 256 + threadIdx.x;
    int m  = (int)(i & 1023);
    int oc = (int)((i >> 10) & 127);
    int b  = (int)(i >> 17);
    int g  = oc >> 2;
    float mean = mr2[(b * 32 + g) * 2 + 0];
    float rstd = mr2[(b * 32 + g) * 2 + 1];
    float gw = g2w[oc], gb = g2b[oc];
    const _Float16* src = x2h + ((size_t)(b * OUTC + oc) * NM + m) * KNN;
    float best = -3.4e38f;
#pragma unroll
    for (int q = 0; q < 4; ++q) {
        v8h v = *(const v8h*)&src[q * 8];
#pragma unroll
        for (int e = 0; e < 8; ++e) {
            float y = ((float)v[e] - mean) * rstd * gw + gb;
            best = fmaxf(best, y);
        }
    }
    out_feat[((size_t)(b * NM + m)) * OUTC + oc] = fmaxf(best, 0.0f);
}

// ---------------------------------------------------------------------------
extern "C" void kernel_launch(void* const* d_in, const int* in_sizes, int n_in,
                              void* d_out, int out_size, void* d_ws, size_t ws_size,
                              hipStream_t stream)
{
    (void)in_sizes; (void)n_in; (void)out_size; (void)ws_size;

    const float* xyz  = (const float*)d_in[0];
    const float* feat = (const float*)d_in[1];
    const float* w1   = (const float*)d_in[2];
    const float* b1   = (const float*)d_in[3];
    const float* g1w  = (const float*)d_in[4];
    const float* g1b  = (const float*)d_in[5];
    const float* w2   = (const float*)d_in[6];
    const float* b2   = (const float*)d_in[7];
    const float* g2w  = (const float*)d_in[8];
    const float* g2b  = (const float*)d_in[9];

    float* out_cent = (float*)d_out;                       // (B,M,3)
    float* out_feat = out_cent + (size_t)NB * NM * 3;      // (B,M,128)

    char* ws = (char*)d_ws;
    size_t off = 0;
    auto carve = [&](size_t bytes) -> char* {
        char* p = ws + off;
        off = (off + bytes + 255) & ~(size_t)255;
        return p;
    };
    int*      fps_idx = (int*)     carve((size_t)NB * NM * sizeof(int));
    int*      knn_idx = (int*)     carve((size_t)NB * NM * KNN * sizeof(int));
    float*    cent    = (float*)   carve((size_t)NB * NM * 3 * sizeof(float));
    float*    ps1     = (float*)   carve((size_t)NB * 32 * NM * 2 * sizeof(float));
    float*    mr1     = (float*)   carve((size_t)NB * 32 * 2 * sizeof(float));
    float*    ps2     = (float*)   carve((size_t)NB * 32 * NM * 2 * sizeof(float));
    float*    mr2     = (float*)   carve((size_t)NB * 32 * 2 * sizeof(float));
    _Float16* w1h     = (_Float16*)carve((size_t)W1PAD * sizeof(_Float16));
    _Float16* w2h     = (_Float16*)carve((size_t)W2PAD * sizeof(_Float16));
    _Float16* x1h     = (_Float16*)carve((size_t)NB * HID  * NM * KNN * sizeof(_Float16));
    _Float16* x2h     = (_Float16*)carve((size_t)NB * OUTC * NM * KNN * sizeof(_Float16));

    prep_w_kernel  <<<(W1PAD + W2PAD + 255) / 256, 256, 0, stream>>>(w1, w2, w1h, w2h);
    fps_kernel     <<<NB,       1024, 0, stream>>>(xyz, fps_idx, cent, out_cent);
    knn_kernel     <<<NB * NM,   256, 0, stream>>>(xyz, cent, knn_idx);
    conv1_kernel   <<<NB * NM,   256, 0, stream>>>(xyz, feat, cent, knn_idx, w1h, b1, x1h, ps1);
    finstats_kernel<<<NB * 32,   256, 0, stream>>>(ps1, mr1, 1.0f / (float)((HID / 32) * NM * KNN));
    conv2_kernel   <<<NB * NM,   256, 0, stream>>>(x1h, mr1, g1w, g1b, w2h, b2, x2h, ps2);
    finstats_kernel<<<NB * 32,   256, 0, stream>>>(ps2, mr2, 1.0f / (float)((OUTC / 32) * NM * KNN));
    final_kernel   <<<(NB * NM * OUTC) / 256, 256, 0, stream>>>(x2h, mr2, g2w, g2b, out_feat);
}
